// EnhancedMeshGraphNetsProcessor_38276748542152
// MI455X (gfx1250) — matrix operand order repaired and checked by reading, not compiled, not run on hardware
//
#include <hip/hip_runtime.h>
#include <hip/hip_bf16.h>
#include <math.h>

// Problem sizes (fixed by the reference)
#define NN 20000
#define EE 200000
#define HH 128
#define LL 10
#define NHEADS 4
#define HD 32

typedef __bf16 bf16_t;
typedef __attribute__((ext_vector_type(16))) __bf16 v16bf;
typedef __attribute__((ext_vector_type(8)))  float  v8f;
typedef __attribute__((ext_vector_type(4)))  float  f32x4;
typedef __attribute__((ext_vector_type(4)))  unsigned int u32x4;   // POD 16B vector (HIP uint4 has ctors)

union FragAB { v16bf v; u32x4 q[2]; };
union PK8    { bf16_t h[8]; u32x4 v; };

__device__ __forceinline__ bf16_t f2bf(float f) { return (bf16_t)f; }

__device__ __forceinline__ float gelu_exact(float x) {
    return 0.5f * x * (1.0f + erff(x * 0.70710678118654752f));
}

// A-fragment (16x32 bf16) from LDS, row-major [16][stride] bf16, 16B-aligned rows.
// lanes 0-15: M=lane, K = 0..15 ; lanes 16-31: M=lane-16, K = 16..31 (matches load_b_frag)
__device__ __forceinline__ v16bf load_a_frag(const bf16_t* base, int stride, int kk, int lane) {
    int m  = lane & 15;
    int k0 = kk * 32 + ((lane >> 4) & 1) * 16;
    FragAB f;
    f.q[0] = *reinterpret_cast<const u32x4*>(base + m * stride + k0);
    f.q[1] = *reinterpret_cast<const u32x4*>(base + m * stride + k0 + 8);
    return f.v;
}

// B-fragment (32x16 bf16) from global weights stored TRANSPOSED: wT[n][k], k contiguous.
// lanes 0-15: N=lane, K = kk*32 + 0..15 ; lanes 16-31: N=lane-16, K = kk*32 + 16..31
__device__ __forceinline__ v16bf load_b_frag(const bf16_t* __restrict__ wT, int K, int n0, int kk, int lane) {
    int n  = n0 + (lane & 15);
    int kb = kk * 32 + ((lane >> 4) & 1) * 16;
    const bf16_t* p = wT + (size_t)n * K + kb;
    FragAB f;
    f.q[0] = *reinterpret_cast<const u32x4*>(p);
    f.q[1] = *reinterpret_cast<const u32x4*>(p + 8);
    return f.v;
}

// C/D f32 16x16: VGPR v holds M = v + 8*(lane>=16), N = lane&15
__device__ __forceinline__ void store_c_frag(float* base, int stride, int n0, v8f c, int lane) {
    int n  = n0 + (lane & 15);
    int m0 = ((lane >> 4) & 1) * 8;
#pragma unroll
    for (int i = 0; i < 8; ++i) base[(m0 + i) * stride + n] = c[i];
}

// ---------------------------------------------------------------------------
// Fused two-layer MLP:  LN2(GELU(LN1(concat3 @ W1 + b1)) @ W2 + b2) + residual
// MODE 0 (edge): concat = [x[row], x[col], edge_attr]; residual/out = edge_attr (in place);
//                fused scatter-add of the result into agg[col] (segment-sum).
// MODE 1 (node): concat = [x, agg/deg, ctx]; residual/out = x (in place).
// One block = 4 wave32 = 32 rows (two 16-row M-tiles sharing every B fragment ->
// halves L2 weight traffic vs a 16-row tile). GEMM1 [32x384]@[384x256],
// GEMM2 [32x256]@[256x128]. LN epilogues run in two 16-row phases so a single
// 16-row f32 staging buffer serves both GEMMs (static LDS ~57.5 KB).
// ---------------------------------------------------------------------------
template<int MODE>
__global__ __launch_bounds__(128) void fused_mlp_kernel(
    const float* xin, const float* ein, const int* __restrict__ eidx,
    const float* __restrict__ agg, const float* __restrict__ deg, const float* __restrict__ ctx,
    const bf16_t* __restrict__ w1T, const float* __restrict__ b1,
    const float* __restrict__ g1, const float* __restrict__ be1,
    const bf16_t* __restrict__ w2T, const float* __restrict__ b2,
    const float* __restrict__ g2, const float* __restrict__ be2,
    float* outp, float* scat)
{
    __shared__ __align__(16) bf16_t sA[32][392];   // A stage, 784B rows (16B aligned)
    __shared__ __align__(16) bf16_t sM[32][264];   // mid bf16 (A of GEMM2)
    __shared__ __align__(16) float  sO[16][264];   // reused: GEMM1-out half / GEMM2-out half

    const int tid  = threadIdx.x;
    const int lane = tid & 31;
    const int wv   = tid >> 5;
    const int r    = tid >> 3;     // 0..15, 8 threads per row
    const int sub  = tid & 7;
    const long base = (long)blockIdx.x * 32;

    int ri[2] = {0, 0}, ci[2] = {0, 0};
    float invd[2] = {1.f, 1.f};
#pragma unroll
    for (int mm = 0; mm < 2; ++mm) {
        long row = base + mm * 16 + r;
        if (MODE == 0) { ri[mm] = eidx[row]; ci[mm] = eidx[EE + row]; }
        if (MODE == 1) invd[mm] = 1.0f / fmaxf(deg[row], 1.0f);
    }

    // ---- stage A: gather + f32->bf16 convert, vectorized (f32x4 loads, b128 LDS stores)
#pragma unroll
    for (int mm = 0; mm < 2; ++mm) {
        const int rr = mm * 16 + r;
        const long row = base + rr;
#pragma unroll
        for (int s = 0; s < 3; ++s) {
            const float* src;
            float scale = 1.0f;
            if (MODE == 0) {
                src = (s == 0) ? xin + (size_t)ri[mm] * HH
                    : (s == 1) ? xin + (size_t)ci[mm] * HH
                               : ein + row * HH;
            } else {
                src = (s == 0) ? xin + row * HH : (s == 1) ? agg + row * HH : ctx + row * HH;
                if (s == 1) scale = invd[mm];
            }
            const f32x4* p = reinterpret_cast<const f32x4*>(src + sub * 16);
#pragma unroll
            for (int g = 0; g < 2; ++g) {
                f32x4 q0 = p[g * 2 + 0];
                f32x4 q1 = p[g * 2 + 1];
                PK8 pk;
#pragma unroll
                for (int i = 0; i < 4; ++i) { pk.h[i] = f2bf(q0[i] * scale); pk.h[4 + i] = f2bf(q1[i] * scale); }
                *reinterpret_cast<u32x4*>(&sA[rr][s * 128 + sub * 16 + g * 8]) = pk.v;
            }
        }
    }
    __syncthreads();

    // ---- GEMM1: K=384 (12 k-steps), Nout=256; each wave: 4 n-tiles x 2 m-tiles
    v8f acc1[2][4] = {};
    for (int kk = 0; kk < 12; ++kk) {
        v16bf a0 = load_a_frag(&sA[0][0],  392, kk, lane);
        v16bf a1 = load_a_frag(&sA[16][0], 392, kk, lane);
#pragma unroll
        for (int j = 0; j < 4; ++j) {
            v16bf b = load_b_frag(w1T, 384, (wv * 4 + j) * 16, kk, lane);
            acc1[0][j] = __builtin_amdgcn_wmma_f32_16x16x32_bf16(false, a0, false, b, (short)0, acc1[0][j], false, false);
            acc1[1][j] = __builtin_amdgcn_wmma_f32_16x16x32_bf16(false, a1, false, b, (short)0, acc1[1][j], false, false);
        }
    }

    // ---- bias + LayerNorm + GELU -> bf16 mid, in two 16-row phases through sO
#pragma unroll
    for (int mm = 0; mm < 2; ++mm) {
        __syncthreads();   // sO free (prev phase consumed / sA reads done)
#pragma unroll
        for (int j = 0; j < 4; ++j) store_c_frag(&sO[0][0], 264, (wv * 4 + j) * 16, acc1[mm][j], lane);
        __syncthreads();
        float s = 0.f, s2 = 0.f;
        const int c0 = sub * 32;
#pragma unroll
        for (int i = 0; i < 32; ++i) {
            int c = c0 + i;
            float v = sO[r][c] + b1[c];
            sO[r][c] = v;
            s += v; s2 += v * v;
        }
#pragma unroll
        for (int m = 4; m >= 1; m >>= 1) { s += __shfl_xor(s, m, 8); s2 += __shfl_xor(s2, m, 8); }
        float mean = s * (1.0f / 256.0f);
        float var  = s2 * (1.0f / 256.0f) - mean * mean;
        float rstd = rsqrtf(var + 1e-5f);
#pragma unroll
        for (int i = 0; i < 32; ++i) {
            int c = c0 + i;
            float v = (sO[r][c] - mean) * rstd * g1[c] + be1[c];
            sM[mm * 16 + r][c] = f2bf(gelu_exact(v));
        }
    }
    __syncthreads();

    // ---- GEMM2: K=256 (8 k-steps), Nout=128; each wave: 2 n-tiles x 2 m-tiles
    v8f acc2[2][2] = {};
    for (int kk = 0; kk < 8; ++kk) {
        v16bf a0 = load_a_frag(&sM[0][0],  264, kk, lane);
        v16bf a1 = load_a_frag(&sM[16][0], 264, kk, lane);
#pragma unroll
        for (int j = 0; j < 2; ++j) {
            v16bf b = load_b_frag(w2T, 256, (wv * 2 + j) * 16, kk, lane);
            acc2[0][j] = __builtin_amdgcn_wmma_f32_16x16x32_bf16(false, a0, false, b, (short)0, acc2[0][j], false, false);
            acc2[1][j] = __builtin_amdgcn_wmma_f32_16x16x32_bf16(false, a1, false, b, (short)0, acc2[1][j], false, false);
        }
    }

    // ---- bias + LayerNorm + residual write (+ fused scatter-add), two phases
#pragma unroll
    for (int mm = 0; mm < 2; ++mm) {
        __syncthreads();   // sO free / prior phase reads done
#pragma unroll
        for (int j = 0; j < 2; ++j) store_c_frag(&sO[0][0], 264, (wv * 2 + j) * 16, acc2[mm][j], lane);
        __syncthreads();
        float s = 0.f, s2 = 0.f;
        const int c0 = sub * 16;
        float vals[16];
#pragma unroll
        for (int i = 0; i < 16; ++i) {
            float v = sO[r][c0 + i] + b2[c0 + i];
            vals[i] = v;
            s += v; s2 += v * v;
        }
#pragma unroll
        for (int m = 4; m >= 1; m >>= 1) { s += __shfl_xor(s, m, 8); s2 += __shfl_xor(s2, m, 8); }
        float mean = s * (1.0f / 128.0f);
        float var  = s2 * (1.0f / 128.0f) - mean * mean;
        float rstd = rsqrtf(var + 1e-5f);
        const long row = base + mm * 16 + r;
        f32x4* op = reinterpret_cast<f32x4*>(outp + row * HH + c0);
        f32x4 res[4];
#pragma unroll
        for (int j = 0; j < 4; ++j) res[j] = op[j];            // residual (in-place buffer)
#pragma unroll
        for (int i = 0; i < 16; ++i) {
            int c = c0 + i;
            res[i >> 2][i & 3] += (vals[i] - mean) * rstd * g2[c] + be2[c];
        }
#pragma unroll
        for (int j = 0; j < 4; ++j) op[j] = res[j];
        if (MODE == 0) {
#pragma unroll
            for (int i = 0; i < 16; ++i)
                atomicAdd(&scat[(size_t)ci[mm] * HH + c0 + i], res[i >> 2][i & 3]);  // segment-sum
        }
    }
}

// ---------------------------------------------------------------------------
// Generic 16-row WMMA projection for the global-token attention, K = 128.
// MODE 0: kv = x@[wk|wv]+b  -> kv[N,256] f32, plus token-attention logits[N,8]
// MODE 1: qn = x@wq_nod+b, then fused softmax over the 2 tokens -> attnA[N,128] bf16
// MODE 2: ctx = attnA@wo_nod + bo -> ctx[N,128] f32
// ---------------------------------------------------------------------------
template<int MODE>
__global__ __launch_bounds__(128) void att_gemm_kernel(
    const float* __restrict__ Af, const bf16_t* __restrict__ Ab,
    const bf16_t* __restrict__ WT, const float* __restrict__ bias,
    float* __restrict__ outf, float* __restrict__ logits,
    const float* __restrict__ qtok, const float* __restrict__ kt, const float* __restrict__ vt,
    bf16_t* __restrict__ attnA)
{
    constexpr int NOUT = (MODE == 0) ? 256 : 128;
    constexpr int TPW  = NOUT / 64;     // n-tiles per wave
    __shared__ __align__(16) bf16_t sA[16][136];
    __shared__ __align__(16) float  sO[16][264];

    const int tid = threadIdx.x, lane = tid & 31, wv = tid >> 5;
    const int r = tid >> 3, sub = tid & 7;
    const long row = (long)blockIdx.x * 16 + r;

    for (int i = 0; i < 16; ++i) {
        int c = sub * 16 + i;
        sA[r][c] = (MODE == 2) ? Ab[row * HH + c] : f2bf(Af[row * HH + c]);
    }
    __syncthreads();

    v8f acc[TPW] = {};
    for (int kk = 0; kk < 4; ++kk) {
        v16bf a = load_a_frag(&sA[0][0], 136, kk, lane);
#pragma unroll
        for (int j = 0; j < TPW; ++j) {
            v16bf b = load_b_frag(WT, 128, (wv * TPW + j) * 16, kk, lane);
            acc[j] = __builtin_amdgcn_wmma_f32_16x16x32_bf16(false, a, false, b, (short)0, acc[j], false, false);
        }
    }
#pragma unroll
    for (int j = 0; j < TPW; ++j) store_c_frag(&sO[0][0], 264, (wv * TPW + j) * 16, acc[j], lane);
    __syncthreads();

    constexpr int CPT = NOUT / 8;
    for (int i = 0; i < CPT; ++i) {
        int c = sub * CPT + i;
        float v = sO[r][c] + bias[c];
        sO[r][c] = v;
        if (MODE == 0) outf[row * 256 + c] = v;
        if (MODE == 2) outf[row * HH + c] = v;
    }
    __syncthreads();

    if (MODE == 0) {
        // logits s[t,h] = q[t,h]·k[n,h] / sqrt(32); pair index = sub (t = sub>>2, h = sub&3)
        int t = sub >> 2, h = sub & 3;
        float s = 0.f;
#pragma unroll
        for (int d = 0; d < 32; ++d) s += qtok[t * HH + h * HD + d] * sO[r][h * HD + d];
        logits[row * 8 + sub] = s * 0.17677669529663687f;
    } else if (MODE == 1) {
        if (sub < 4) {
            int h = sub;
            float s0 = 0.f, s1 = 0.f;
#pragma unroll
            for (int d = 0; d < 32; ++d) {
                float qv = sO[r][h * HD + d];
                s0 += qv * kt[h * HD + d];
                s1 += qv * kt[HH + h * HD + d];
            }
            s0 *= 0.17677669529663687f; s1 *= 0.17677669529663687f;
            float mx = fmaxf(s0, s1);
            float e0 = expf(s0 - mx), e1 = expf(s1 - mx);
            float inv = 1.0f / (e0 + e1);
#pragma unroll
            for (int d = 0; d < 32; ++d) {
                float o = e0 * inv * vt[h * HD + d] + e1 * inv * vt[HH + h * HD + d];
                attnA[row * HH + h * HD + d] = f2bf(o);
            }
        }
    }
}

// ---- one-time weight convert + transpose: dst[n*K + k] = bf16(src[k*stride + colOff + n])
__global__ void convertT_kernel(const float* __restrict__ src, bf16_t* __restrict__ dst,
                                int K, int Nt, int srcStride, int colOff) {
    int i = blockIdx.x * 256 + threadIdx.x;
    if (i >= K * Nt) return;
    int n = i / K, k = i - n * K;
    dst[i] = f2bf(src[(size_t)k * srcStride + colOff + n]);
}

__global__ void deg_kernel(const int* __restrict__ eidx, float* __restrict__ degb) {
    int e = blockIdx.x * 256 + threadIdx.x;
    if (e < EE) atomicAdd(&degb[eidx[EE + e]], 1.0f);
}

// q = token_embed @ wq + bq  (layer-invariant; 2x128 outputs)
__global__ void tokens_q_kernel(const float* __restrict__ tok, const float* __restrict__ wqkv,
                                const float* __restrict__ bqkv, float* __restrict__ q) {
    int tid = threadIdx.x; int t = tid >> 7, j = tid & 127;
    float a = bqkv[j];
    for (int k = 0; k < 128; ++k) a += tok[t * 128 + k] * wqkv[k * 384 + j];
    q[t * 128 + j] = a;
}

__global__ void logit_max_p1(const float* __restrict__ logits, float* __restrict__ pmax) {
    __shared__ float sm[256][8];
    int tid = threadIdx.x;
    float m[8];
#pragma unroll
    for (int j = 0; j < 8; ++j) m[j] = -1e30f;
    for (int i = blockIdx.x * 256 + tid; i < NN; i += 256 * 256) {
#pragma unroll
        for (int j = 0; j < 8; ++j) m[j] = fmaxf(m[j], logits[(size_t)i * 8 + j]);
    }
#pragma unroll
    for (int j = 0; j < 8; ++j) sm[tid][j] = m[j];
    __syncthreads();
    for (int s = 128; s >= 1; s >>= 1) {
        if (tid < s) {
#pragma unroll
            for (int j = 0; j < 8; ++j) sm[tid][j] = fmaxf(sm[tid][j], sm[tid + s][j]);
        }
        __syncthreads();
    }
    if (tid < 8) pmax[blockIdx.x * 8 + tid] = sm[0][tid];
}

__global__ void logit_max_p2(const float* __restrict__ pmax, float* __restrict__ gmax) {
    int tid = threadIdx.x;
    if (tid < 8) {
        float m = -1e30f;
        for (int b = 0; b < 256; ++b) m = fmaxf(m, pmax[b * 8 + tid]);
        gmax[tid] = m;
    }
}

// thread = (pair j, dim d); partial softmax numerator sums over nodes
__global__ void attn_sum_p1(const float* __restrict__ logits, const float* __restrict__ kv,
                            const float* __restrict__ gmax, float* __restrict__ psum,
                            float* __restrict__ pov) {
    int tid = threadIdx.x, j = tid >> 5, d = tid & 31, h = j & 3;
    float gm = gmax[j], se = 0.f, ov = 0.f;
    for (int i = blockIdx.x; i < NN; i += 256) {
        float e = expf(logits[(size_t)i * 8 + j] - gm);
        se += e;
        ov += e * kv[(size_t)i * 256 + 128 + h * HD + d];
    }
    pov[(blockIdx.x * 8 + j) * 32 + d] = ov;
    if (d == 0) psum[blockIdx.x * 8 + j] = se;
}

__global__ void attn_sum_p2(const float* __restrict__ psum, const float* __restrict__ pov,
                            float* __restrict__ attn_out) {
    __shared__ float ssum[8];
    int tid = threadIdx.x, j = tid >> 5, d = tid & 31;
    if (tid < 8) {
        float s = 0.f;
        for (int b = 0; b < 256; ++b) s += psum[b * 8 + tid];
        ssum[tid] = s;
    }
    __syncthreads();
    float o = 0.f;
    for (int b = 0; b < 256; ++b) o += pov[(b * 8 + j) * 32 + d];
    int t = j >> 2, h = j & 3;
    attn_out[t * 128 + h * HD + d] = o / ssum[j];
}

// tokens: out-proj + LN1 + FF + LN2, then k_t/v_t for node attention. Tiny, 1 block.
__global__ void tokens_update_kernel(
    const float* __restrict__ attn_out, const float* __restrict__ tok,
    const float* __restrict__ wo, const float* __restrict__ bo,
    const float* __restrict__ g1, const float* __restrict__ b1,
    const float* __restrict__ fw1, const float* __restrict__ fb1,
    const float* __restrict__ fw2, const float* __restrict__ fb2,
    const float* __restrict__ g2, const float* __restrict__ b2,
    const float* __restrict__ nwqkv, const float* __restrict__ nbqkv,
    float* __restrict__ kt, float* __restrict__ vt)
{
    __shared__ float s1[2][128], sh[2][512], s2[2][128];
    __shared__ float mu[2], rs[2];
    int tid = threadIdx.x, t = tid >> 7, j = tid & 127;

    float a = bo[j];
    for (int k = 0; k < 128; ++k) a += attn_out[t * 128 + k] * wo[k * 128 + j];
    s1[t][j] = tok[t * 128 + j] + a;
    __syncthreads();
    if (j == 0) {
        float s = 0.f, ss = 0.f;
        for (int k = 0; k < 128; ++k) { float v = s1[t][k]; s += v; ss += v * v; }
        float m = s * (1.f / 128.f); mu[t] = m; rs[t] = rsqrtf(ss * (1.f / 128.f) - m * m + 1e-5f);
    }
    __syncthreads();
    float v1 = (s1[t][j] - mu[t]) * rs[t] * g1[j] + b1[j];
    s1[t][j] = v1;
    __syncthreads();
    for (int o = tid; o < 1024; o += 256) {
        int tt = o >> 9, c = o & 511;
        float acc = fb1[c];
        for (int k = 0; k < 128; ++k) acc += s1[tt][k] * fw1[k * 512 + c];
        sh[tt][c] = gelu_exact(acc);
    }
    __syncthreads();
    float a2 = fb2[j];
    for (int k = 0; k < 512; ++k) a2 += sh[t][k] * fw2[k * 128 + j];
    s2[t][j] = s1[t][j] + a2;
    __syncthreads();
    if (j == 0) {
        float s = 0.f, ss = 0.f;
        for (int k = 0; k < 128; ++k) { float v = s2[t][k]; s += v; ss += v * v; }
        float m = s * (1.f / 128.f); mu[t] = m; rs[t] = rsqrtf(ss * (1.f / 128.f) - m * m + 1e-5f);
    }
    __syncthreads();
    float v2 = (s2[t][j] - mu[t]) * rs[t] * g2[j] + b2[j];
    s2[t][j] = v2;
    __syncthreads();
    float ka = nbqkv[128 + j], va = nbqkv[256 + j];
    for (int k = 0; k < 128; ++k) {
        float tv = s2[t][k];
        ka += tv * nwqkv[k * 384 + 128 + j];
        va += tv * nwqkv[k * 384 + 256 + j];
    }
    kt[t * 128 + j] = ka;
    vt[t * 128 + j] = va;
}

// ---------------------------------------------------------------------------
extern "C" void kernel_launch(void* const* d_in, const int* in_sizes, int n_in,
                              void* d_out, int out_size, void* d_ws, size_t ws_size,
                              hipStream_t stream) {
    (void)in_sizes; (void)n_in; (void)out_size; (void)ws_size;
    const float* x0    = (const float*)d_in[0];
    const int*   eidx  = (const int*)d_in[1];
    const float* ea0   = (const float*)d_in[2];
    const float* ew1   = (const float*)d_in[3];
    const float* eb1   = (const float*)d_in[4];
    const float* elg1  = (const float*)d_in[5];
    const float* elb1  = (const float*)d_in[6];
    const float* ew2   = (const float*)d_in[7];
    const float* eb2   = (const float*)d_in[8];
    const float* elg2  = (const float*)d_in[9];
    const float* elb2  = (const float*)d_in[10];
    const float* nw1   = (const float*)d_in[11];
    const float* nb1   = (const float*)d_in[12];
    const float* nlg1  = (const float*)d_in[13];
    const float* nlb1  = (const float*)d_in[14];
    const float* nw2   = (const float*)d_in[15];
    const float* nb2   = (const float*)d_in[16];
    const float* nlg2  = (const float*)d_in[17];
    const float* nlb2  = (const float*)d_in[18];
    const float* tokE  = (const float*)d_in[19];
    const float* twqkv = (const float*)d_in[20];
    const float* tbqkv = (const float*)d_in[21];
    const float* two   = (const float*)d_in[22];
    const float* tbo   = (const float*)d_in[23];
    const float* lt1g  = (const float*)d_in[24];
    const float* lt1b  = (const float*)d_in[25];
    const float* lt2g  = (const float*)d_in[26];
    const float* lt2b  = (const float*)d_in[27];
    const float* fw1   = (const float*)d_in[28];
    const float* fb1   = (const float*)d_in[29];
    const float* fw2   = (const float*)d_in[30];
    const float* fb2   = (const float*)d_in[31];
    const float* nqkv  = (const float*)d_in[32];
    const float* nbq   = (const float*)d_in[33];
    const float* nwo   = (const float*)d_in[34];
    const float* nbo   = (const float*)d_in[35];

    // working x / edge_attr live in d_out (updated in place; final state == output)
    float* xbuf = (float*)d_out;
    float* ebuf = xbuf + (size_t)NN * HH;

    char* ws = (char*)d_ws;
    size_t off = 0;
    auto take = [&](size_t bytes) -> void* {
        void* p = (void*)(ws + off);
        off += (bytes + 255) & ~(size_t)255;
        return p;
    };
    bf16_t* ew1T  = (bf16_t*)take((size_t)LL * 256 * 384 * sizeof(bf16_t));
    bf16_t* ew2T  = (bf16_t*)take((size_t)LL * 128 * 256 * sizeof(bf16_t));
    bf16_t* nw1T  = (bf16_t*)take((size_t)LL * 256 * 384 * sizeof(bf16_t));
    bf16_t* nw2T  = (bf16_t*)take((size_t)LL * 128 * 256 * sizeof(bf16_t));
    bf16_t* kvT   = (bf16_t*)take(256 * 128 * sizeof(bf16_t));
    bf16_t* wqnT  = (bf16_t*)take(128 * 128 * sizeof(bf16_t));
    bf16_t* wonT  = (bf16_t*)take(128 * 128 * sizeof(bf16_t));
    float*  qtok  = (float*)take(2 * 128 * sizeof(float));
    float*  kv    = (float*)take((size_t)NN * 256 * sizeof(float));
    float*  logit = (float*)take((size_t)NN * 8 * sizeof(float));
    float*  pmax  = (float*)take(256 * 8 * sizeof(float));
    float*  gmax  = (float*)take(8 * sizeof(float));
    float*  psum  = (float*)take(256 * 8 * sizeof(float));
    float*  pov   = (float*)take(256 * 8 * 32 * sizeof(float));
    float*  aout  = (float*)take(256 * sizeof(float));
    float*  ktb   = (float*)take(256 * sizeof(float));
    float*  vtb   = (float*)take(256 * sizeof(float));
    bf16_t* attnA = (bf16_t*)take((size_t)NN * HH * sizeof(bf16_t));
    float*  ctx   = (float*)take((size_t)NN * HH * sizeof(float));
    float*  agg   = (float*)take((size_t)NN * HH * sizeof(float));
    float*  degb  = (float*)take((size_t)NN * sizeof(float));

    hipMemcpyAsync(xbuf, x0,  (size_t)NN * HH * sizeof(float), hipMemcpyDeviceToDevice, stream);
    hipMemcpyAsync(ebuf, ea0, (size_t)EE * HH * sizeof(float), hipMemcpyDeviceToDevice, stream);

    auto convT = [&](const float* src, bf16_t* dst, int K, int Nt, int stride, int colOff) {
        int tot = K * Nt;
        convertT_kernel<<<(tot + 255) / 256, 256, 0, stream>>>(src, dst, K, Nt, stride, colOff);
    };
    for (int l = 0; l < LL; ++l) {
        convT(ew1 + (size_t)l * 384 * 256, ew1T + (size_t)l * 256 * 384, 384, 256, 256, 0);
        convT(ew2 + (size_t)l * 256 * 128, ew2T + (size_t)l * 128 * 256, 256, 128, 128, 0);
        convT(nw1 + (size_t)l * 384 * 256, nw1T + (size_t)l * 256 * 384, 384, 256, 256, 0);
        convT(nw2 + (size_t)l * 256 * 128, nw2T + (size_t)l * 128 * 256, 256, 128, 128, 0);
    }
    convT(twqkv, kvT, 128, 256, 384, 128);  // [wk|wv] transposed
    convT(nqkv,  wqnT, 128, 128, 384, 0);   // node wq transposed
    convT(nwo,   wonT, 128, 128, 128, 0);   // node wo transposed

    hipMemsetAsync(degb, 0, (size_t)NN * sizeof(float), stream);
    deg_kernel<<<(EE + 255) / 256, 256, 0, stream>>>(eidx, degb);
    tokens_q_kernel<<<1, 256, 0, stream>>>(tokE, twqkv, tbqkv, qtok);

    for (int l = 0; l < LL; ++l) {
        hipMemsetAsync(agg, 0, (size_t)NN * HH * sizeof(float), stream);
        // edge MLP (WMMA) + fused scatter-add          E/32 = 6250 tiles
        fused_mlp_kernel<0><<<EE / 32, 128, 0, stream>>>(
            xbuf, ebuf, eidx, nullptr, nullptr, nullptr,
            ew1T + (size_t)l * 256 * 384, eb1 + l * 256, elg1 + l * 256, elb1 + l * 256,
            ew2T + (size_t)l * 128 * 256, eb2 + l * 128, elg2 + l * 128, elb2 + l * 128,
            ebuf, agg);
        // global token context
        att_gemm_kernel<0><<<NN / 16, 128, 0, stream>>>(
            xbuf, nullptr, kvT, tbqkv + 128, kv, logit, qtok, nullptr, nullptr, nullptr);
        logit_max_p1<<<256, 256, 0, stream>>>(logit, pmax);
        logit_max_p2<<<1, 256, 0, stream>>>(pmax, gmax);
        attn_sum_p1<<<256, 256, 0, stream>>>(logit, kv, gmax, psum, pov);
        attn_sum_p2<<<1, 256, 0, stream>>>(psum, pov, aout);
        tokens_update_kernel<<<1, 256, 0, stream>>>(
            aout, tokE, two, tbo, lt1g, lt1b, fw1, fb1, fw2, fb2, lt2g, lt2b, nqkv, nbq, ktb, vtb);
        att_gemm_kernel<1><<<NN / 16, 128, 0, stream>>>(
            xbuf, nullptr, wqnT, nbq, nullptr, nullptr, nullptr, ktb, vtb, attnA);
        att_gemm_kernel<2><<<NN / 16, 128, 0, stream>>>(
            nullptr, attnA, wonT, nbo, ctx, nullptr, nullptr, nullptr, nullptr, nullptr);
        // node MLP (WMMA), in-place update of x             N/32 = 625 tiles
        fused_mlp_kernel<1><<<NN / 32, 128, 0, stream>>>(
            xbuf, nullptr, nullptr, agg, degb, ctx,
            nw1T + (size_t)l * 256 * 384, nb1 + l * 256, nlg1 + l * 256, nlb1 + l * 256,
            nw2T + (size_t)l * 128 * 256, nb2 + l * 128, nlg2 + l * 128, nlb2 + l * 128,
            xbuf, nullptr);
    }
}
